// Attention_53025666236428
// MI455X (gfx1250) — matrix operand order repaired and checked by reading, not compiled, not run on hardware
//
#include <hip/hip_runtime.h>
#include <hip/hip_bf16.h>
#include <stdint.h>

typedef _Float16 half_t;
typedef _Float16 v16h __attribute__((ext_vector_type(16)));
typedef _Float16 v8h  __attribute__((ext_vector_type(8)));
typedef float    v8f  __attribute__((ext_vector_type(8)));

#define DEV static __device__ __forceinline__

DEV v16h cat8(v8h lo, v8h hi) {
    return __builtin_shufflevector(lo, hi, 0,1,2,3,4,5,6,7,8,9,10,11,12,13,14,15);
}
// two 16-byte loads -> one 32-byte WMMA operand
DEV v16h ldpair(const half_t* p0, const half_t* p1) {
    return cat8(*(const v8h*)p0, *(const v8h*)p1);
}
DEV v8f wmma_f16(v16h a, v16h b, v8f c) {
    return __builtin_amdgcn_wmma_f32_16x16x32_f16(false, a, false, b, (short)0, c, false, false);
}
// gfx1250 async global->LDS copy: 16B per lane, tracked by ASYNCcnt
DEV void async_g2l_b128(half_t* lds_dst, const half_t* gsrc) {
    unsigned int laddr = (unsigned int)(uintptr_t)lds_dst;   // low 32 bits = LDS offset
    asm volatile("global_load_async_to_lds_b128 %0, %1, off"
                 :: "v"(laddr), "v"(gsrc) : "memory");
}
DEV void wait_async0() { asm volatile("s_wait_asynccnt 0" ::: "memory"); }
// keep load group ahead of WMMA group so waitcnt pass emits partial waits
DEV void sched_fence() { __builtin_amdgcn_sched_barrier(0); }

#define DIMD    1024
#define LSEQ    2048
#define NB      2
#define NH      16
#define HD      64
#define ROWS    (NB * LSEQ)          // 4096 GEMM rows
#define NEG_INF (-1e30f)

// ---------- precision conversion ----------
__global__ void k_cvt(const float* __restrict__ x, half_t* __restrict__ xh, int n) {
    int i = blockIdx.x * 256 + threadIdx.x;
    if (i < n) xh[i] = (half_t)x[i];
}
// ---------- LDS-tiled transpose: WT[c][r] = (f16)W[r][c]; coalesced both sides ----------
// grid (cols/32, rows/32), block 256 (= 32x8)
__global__ void k_tr(const float* __restrict__ W, half_t* __restrict__ WT, int rows, int cols) {
    __shared__ half_t tile[32][33];
    const int tx = threadIdx.x & 31, ty = threadIdx.x >> 5;
    const int bx = blockIdx.x * 32;   // col base in W
    const int by = blockIdx.y * 32;   // row base in W
#pragma unroll
    for (int i = 0; i < 32; i += 8)
        tile[ty + i][tx] = (half_t)W[(size_t)(by + ty + i) * cols + bx + tx];
    __syncthreads();
#pragma unroll
    for (int i = 0; i < 32; i += 8)
        WT[(size_t)(bx + ty + i) * rows + by + tx] = tile[tx][ty + i];
}

// ---------- stage 1: qkv = x @ Wqkv, scatter into Q / K / Vt ----------
// grid (ROWS/16, 3072/512), block 256 (8 waves); each wave: 16x64 tile.
// A tile (16x1024, 32KB) staged once per block into LDS via async copy.
__global__ void k_qkv(const half_t* __restrict__ xh, const half_t* __restrict__ WT,
                      half_t* __restrict__ Q, half_t* __restrict__ K, half_t* __restrict__ V) {
    __shared__ half_t As[16 * DIMD];
    const int lane = threadIdx.x & 31, wave = threadIdx.x >> 5;
    const int lo = lane & 15, hi = lane >> 4;
    const int r0 = blockIdx.x * 16;
    const int e0 = blockIdx.y * 512 + wave * 64;

    // cooperative async stage of 16 full rows of xh (flat 32KB copy)
    {
        const half_t* g = xh + (size_t)r0 * DIMD;
#pragma unroll
        for (int it = 0; it < 8; ++it) {
            int off = (it * 256 + threadIdx.x) * 8;          // half index, 16B chunks
            async_g2l_b128(As + off, g + off);
        }
        wait_async0();
        __syncthreads();
    }

    v8f acc[4] = {};
    const half_t* arow = As + lo * DIMD + 16 * hi;           // A operand from LDS
    const half_t* brow[4];
#pragma unroll
    for (int t = 0; t < 4; ++t) brow[t] = WT + (e0 + 16 * t + lo) * DIMD + 16 * hi;

    // k-slab of 64: issue ALL loads, fence, then 8 WMMAs (progressive partial waits)
    for (int kk = 0; kk < DIMD; kk += 64) {
        v16h b0[4], b1[4];
#pragma unroll
        for (int t = 0; t < 4; ++t) {
            b0[t] = ldpair(brow[t] + kk,      brow[t] + kk + 8);
            b1[t] = ldpair(brow[t] + kk + 32, brow[t] + kk + 40);
        }
        v16h a0 = ldpair(arow + kk,      arow + kk + 8);
        v16h a1 = ldpair(arow + kk + 32, arow + kk + 40);
        sched_fence();
#pragma unroll
        for (int t = 0; t < 4; ++t) acc[t] = wmma_f16(a0, b0[t], acc[t]);
#pragma unroll
        for (int t = 0; t < 4; ++t) acc[t] = wmma_f16(a1, b1[t], acc[t]);
    }

    // scrambled-reshape scatter: whole 64-wide wave tile lands in one segment
    const int seg = (e0 % 192) / 64;        // 0=Q 1=K 2=V
    const int q192 = e0 / 192;
#pragma unroll
    for (int t = 0; t < 4; ++t) {
        int c = 16 * t + lo;                // 0..63 within segment
#pragma unroll
        for (int j = 0; j < 8; ++j) {
            int l  = r0 + j + 8 * hi;
            int b_ = l >> 11, lr = l & (LSEQ - 1);
            int h  = lr >> 7;
            int l2 = ((lr & 127) << 4) + q192;
            int bh = b_ * NH + h;
            half_t val = (half_t)acc[t][j];
            if (seg == 0)      Q[(bh * LSEQ + l2) * HD + c] = val;
            else if (seg == 1) K[(bh * LSEQ + l2) * HD + c] = val;
            else               V[(bh * HD + c) * LSEQ + l2] = val;   // transposed V
        }
    }
}

// ---------- stage 2: causal flash attention per (b,h), scatter to attnout ----------
// grid (LSEQ/128, NB*NH), block 256 (8 waves); each wave owns 16 query rows
__global__ void k_attn(const half_t* __restrict__ Q, const half_t* __restrict__ K,
                       const half_t* __restrict__ Vt, half_t* __restrict__ AO) {
    __shared__ half_t lds[8 * 16 * 32];     // per-wave 16x32 P tile
    const int lane = threadIdx.x & 31, wave = threadIdx.x >> 5;
    const int lo = lane & 15, hi = lane >> 4;
    const int bh = blockIdx.y;
    const int r0 = (blockIdx.x * 8 + wave) * 16;
    const half_t* Qb = Q  + (size_t)bh * LSEQ * HD;
    const half_t* Kb = K  + (size_t)bh * LSEQ * HD;
    const half_t* Vb = Vt + (size_t)bh * HD * LSEQ;
    half_t* P = lds + wave * 512;

    const half_t* qrow = Qb + (r0 + lo) * HD + 16 * hi;
    v16h aq0 = ldpair(qrow,      qrow + 8);
    v16h aq1 = ldpair(qrow + 32, qrow + 40);

    float m[8], lsum[8];
    v8f acc[4] = {};
#pragma unroll
    for (int j = 0; j < 8; ++j) { m[j] = NEG_INF; lsum[j] = 0.f; }

    const int nblk = (r0 + 15) / 32 + 1;    // causal: only K-blocks with col <= max row
    for (int kb = 0; kb < nblk; ++kb) {
        const int kbase = kb * 32;
        // issue ALL of this block's global loads up front:
        // K-tiles feed the S-WMMAs now; V-tiles are consumed only after softmax,
        // so their latency hides under S-WMMA + softmax + P-transpose.
        v16h bk[2][2], bv[4];
#pragma unroll
        for (int t = 0; t < 2; ++t) {
            const half_t* krow = Kb + (kbase + 16 * t + lo) * HD + 16 * hi;
            bk[t][0] = ldpair(krow,      krow + 8);
            bk[t][1] = ldpair(krow + 32, krow + 40);
        }
#pragma unroll
        for (int nt = 0; nt < 4; ++nt) {
            const half_t* vrow = Vb + (16 * nt + lo) * LSEQ + kbase + 16 * hi;
            bv[nt] = ldpair(vrow, vrow + 8);
        }
        sched_fence();

        float st[2][8];
#pragma unroll
        for (int t = 0; t < 2; ++t) {
            v8f c = {};
            c = wmma_f16(aq0, bk[t][0], c);
            c = wmma_f16(aq1, bk[t][1], c);
            int col = kbase + 16 * t + lo;
#pragma unroll
            for (int j = 0; j < 8; ++j) {
                int row = r0 + j + 8 * hi;
                st[t][j] = (col <= row) ? c[j] * 0.25f : NEG_INF;   // scale = 1/sqrt(H)
            }
        }
        // online softmax (row j of this half-wave lives across its 16 lanes)
#pragma unroll
        for (int j = 0; j < 8; ++j) {
            float rm = fmaxf(st[0][j], st[1][j]);
            rm = fmaxf(rm, __shfl_xor(rm, 1)); rm = fmaxf(rm, __shfl_xor(rm, 2));
            rm = fmaxf(rm, __shfl_xor(rm, 4)); rm = fmaxf(rm, __shfl_xor(rm, 8));
            float mn = fmaxf(m[j], rm);
            float al = __expf(m[j] - mn);
            m[j] = mn;
            float p0 = __expf(st[0][j] - mn);
            float p1 = __expf(st[1][j] - mn);
            float rs = p0 + p1;
            rs += __shfl_xor(rs, 1); rs += __shfl_xor(rs, 2);
            rs += __shfl_xor(rs, 4); rs += __shfl_xor(rs, 8);
            lsum[j] = lsum[j] * al + rs;
#pragma unroll
            for (int nt = 0; nt < 4; ++nt) acc[nt][j] *= al;
            // C-layout -> row-major 16x32 P tile in LDS
            P[(j + 8 * hi) * 32 + lo]      = (half_t)p0;
            P[(j + 8 * hi) * 32 + 16 + lo] = (half_t)p1;
        }
        asm volatile("s_wait_dscnt 0" ::: "memory");   // LDS in-order per wave; be safe
        const half_t* prow = P + lo * 32 + 16 * hi;
        v16h pa = ldpair(prow, prow + 8);              // A-layout read-back
        asm volatile("" ::: "memory");
#pragma unroll
        for (int nt = 0; nt < 4; ++nt)
            acc[nt] = wmma_f16(pa, bv[nt], acc[nt]);
    }
    // epilogue: 1/l scaling + inverse scrambled reshape into attnout [B, L, D]
    const int b_ = bh >> 4, h = bh & 15;
#pragma unroll
    for (int j = 0; j < 8; ++j) {
        float inv = 1.0f / lsum[j];
        int l2 = r0 + j + 8 * hi;
        int lA = h * 128 + (l2 >> 4);
        int cA = (l2 & 15) * 64;
#pragma unroll
        for (int nt = 0; nt < 4; ++nt)
            AO[((size_t)b_ * LSEQ + lA) * DIMD + cA + 16 * nt + lo] = (half_t)(acc[nt][j] * inv);
    }
}

// ---------- stage 3: out = attnout @ Wo (fp32 result) ----------
// grid (ROWS/16, 1024/512), block 256; same async-A + k-slab scheme
__global__ void k_out(const half_t* __restrict__ A, const half_t* __restrict__ WT,
                      float* __restrict__ out) {
    __shared__ half_t As[16 * DIMD];
    const int lane = threadIdx.x & 31, wave = threadIdx.x >> 5;
    const int lo = lane & 15, hi = lane >> 4;
    const int r0 = blockIdx.x * 16;
    const int e0 = blockIdx.y * 512 + wave * 64;

    {
        const half_t* g = A + (size_t)r0 * DIMD;
#pragma unroll
        for (int it = 0; it < 8; ++it) {
            int off = (it * 256 + threadIdx.x) * 8;
            async_g2l_b128(As + off, g + off);
        }
        wait_async0();
        __syncthreads();
    }

    v8f acc[4] = {};
    const half_t* arow = As + lo * DIMD + 16 * hi;
    const half_t* brow[4];
#pragma unroll
    for (int t = 0; t < 4; ++t) brow[t] = WT + (e0 + 16 * t + lo) * DIMD + 16 * hi;

    for (int kk = 0; kk < DIMD; kk += 64) {
        v16h b0[4], b1[4];
#pragma unroll
        for (int t = 0; t < 4; ++t) {
            b0[t] = ldpair(brow[t] + kk,      brow[t] + kk + 8);
            b1[t] = ldpair(brow[t] + kk + 32, brow[t] + kk + 40);
        }
        v16h a0 = ldpair(arow + kk,      arow + kk + 8);
        v16h a1 = ldpair(arow + kk + 32, arow + kk + 40);
        sched_fence();
#pragma unroll
        for (int t = 0; t < 4; ++t) acc[t] = wmma_f16(a0, b0[t], acc[t]);
#pragma unroll
        for (int t = 0; t < 4; ++t) acc[t] = wmma_f16(a1, b1[t], acc[t]);
    }

#pragma unroll
    for (int t = 0; t < 4; ++t)
#pragma unroll
        for (int j = 0; j < 8; ++j)
            out[(size_t)(r0 + j + 8 * hi) * DIMD + e0 + 16 * t + lo] = acc[t][j];
}

extern "C" void kernel_launch(void* const* d_in, const int* in_sizes, int n_in,
                              void* d_out, int out_size, void* d_ws, size_t ws_size,
                              hipStream_t stream) {
    const float* x    = (const float*)d_in[0];
    // d_in[1] = mask (causal in scrambled coords -> computed arithmetically, unused)
    const float* Wqkv = (const float*)d_in[2];
    const float* Wo   = (const float*)d_in[3];
    float* out = (float*)d_out;

    // workspace layout (f16), 48 MB total
    half_t* xh    = (half_t*)d_ws;                     // 4096*1024
    half_t* WqkvT = xh    + (size_t)ROWS * DIMD;       // 3072*1024
    half_t* WoT   = WqkvT + (size_t)3 * DIMD * DIMD;   // 1024*1024
    half_t* Qm    = WoT   + (size_t)DIMD * DIMD;       // 32*2048*64
    half_t* Km    = Qm    + (size_t)NB * NH * LSEQ * HD;
    half_t* Vt    = Km    + (size_t)NB * NH * LSEQ * HD;
    half_t* AO    = Vt    + (size_t)NB * NH * LSEQ * HD;

    k_cvt<<<(ROWS * DIMD) / 256, 256, 0, stream>>>(x, xh, ROWS * DIMD);
    k_tr <<<dim3((3 * DIMD) / 32, DIMD / 32), 256, 0, stream>>>(Wqkv, WqkvT, DIMD, 3 * DIMD);
    k_tr <<<dim3(DIMD / 32, DIMD / 32),       256, 0, stream>>>(Wo, WoT, DIMD, DIMD);

    k_qkv <<<dim3(ROWS / 16, (3 * DIMD) / 512), 256, 0, stream>>>(xh, WqkvT, Qm, Km, Vt);
    k_attn<<<dim3(LSEQ / 128, NB * NH),        256, 0, stream>>>(Qm, Km, Vt, AO);
    k_out <<<dim3(ROWS / 16, DIMD / 512),      256, 0, stream>>>(AO, WoT, out);
}
